// PageRank_764504178708
// MI455X (gfx1250) — matrix operand order repaired and checked by reading, compiled, not run on hardware
//
#include <hip/hip_runtime.h>
#include <math.h>

#define NN  6144
#define HD  256
#define BM  32
#define BK  64       // K-tile for the small GEMMs (k_proj / k_hw)
#define LDT 72       // padded LDS stride for small GEMMs (16B-aligned rows)
#define BK2 32       // K-tile for the fused big GEMM (double-buffered)
#define LDT2 48      // padded LDS stride (96B rows -> 16B aligned)

typedef __attribute__((ext_vector_type(16))) __bf16 v16bf;
typedef __attribute__((ext_vector_type(8)))  __bf16 v8bf;
typedef __attribute__((ext_vector_type(4)))  __bf16 v4bf;
typedef __attribute__((ext_vector_type(8)))  float  v8f;
typedef __attribute__((vector_size(16)))     int    v4i;

#if __has_builtin(__builtin_amdgcn_global_load_async_to_lds_b128) && \
    __has_builtin(__builtin_amdgcn_s_wait_asynccnt)
#define USE_ASYNC 1
// builtin signature (from hipcc diagnostic): (addrspace(1) v4i*, addrspace(3) v4i*, imm, imm)
#define GPTR(p) ((__attribute__((address_space(1))) v4i*)(p))
#define LPTR(p) ((__attribute__((address_space(3))) v4i*)(p))
#else
#define USE_ASYNC 0
#endif

#define WMMA_BF16(a, b, c) \
  __builtin_amdgcn_wmma_f32_16x16x32_bf16(false, (a), false, (b), (short)0, (c), false, false)

// A fragment (16x32 bf16), LDS tile row-major with stride LD.
// lane l (half = l>>4, m = l&15): K runs [8h, 8h+8) and [16+8h, 16+8h+8)
template <int LD>
static __device__ __forceinline__ v16bf frag_a(const __bf16* base, int lane) {
  const int m = lane & 15, h = lane >> 4;
  const __bf16* p = base + m * LD + 8 * h;
  v8bf lo = *(const v8bf*)p;
  v8bf hi = *(const v8bf*)(p + 16);
  return __builtin_shufflevector(lo, hi, 0,1,2,3,4,5,6,7,8,9,10,11,12,13,14,15);
}

// B fragment (32x16 bf16), LDS tile stored N-major [n][k] with stride LD.
// lane l (half = l>>4, n = l&15): K runs [8h, 8h+8) and [16+8h, 16+8h+8)
template <int LD>
static __device__ __forceinline__ v16bf frag_b(const __bf16* base, int lane) {
  const int n = lane & 15, h = lane >> 4;
  const __bf16* p = base + n * LD + 8 * h;
  v8bf lo = *(const v8bf*)p;
  v8bf hi = *(const v8bf*)(p + 16);
  return __builtin_shufflevector(lo, hi, 0,1,2,3,4,5,6,7,8,9,10,11,12,13,14,15);
}

// ---------------- Kernel 0: transpose + convert weights to bf16 [N][K] ----------------
__global__ __launch_bounds__(256)
void k_prep(const float* __restrict__ Wp, const float* __restrict__ Wg,
            __bf16* __restrict__ WTp, __bf16* __restrict__ WTg) {
  const int i = blockIdx.x * 256 + threadIdx.x;   // i = k*256 + n (row-major read)
  const int k = i >> 8, n = i & 255;
  WTp[n * HD + k] = (__bf16)Wp[i];
  WTg[n * HD + k] = (__bf16)Wg[i];
}

// ---------------- Kernel 1: h2 = tanh(h @ W_proj + b_proj), bf16 out ----------------
__global__ __launch_bounds__(256)
void k_proj(const float* __restrict__ h, const __bf16* __restrict__ WT,
            const float* __restrict__ bias, __bf16* __restrict__ h2) {
  __shared__ __bf16 As[BM][LDT];
  __shared__ __bf16 Bt[HD][LDT];
  const int tid = threadIdx.x, lane = tid & 31, wave = tid >> 5;
  const int row0 = blockIdx.x * BM;
  v8f acc[2][2] = {};

  for (int k0 = 0; k0 < HD; k0 += BK) {
    { // stage A: f32 -> bf16, 8 elems/thread
      const int r = tid >> 3, c = (tid & 7) * 8;
      const float* ap = h + (size_t)(row0 + r) * HD + k0 + c;
      v8bf v;
      #pragma unroll
      for (int j = 0; j < 8; ++j) v[j] = (__bf16)ap[j];
      *(v8bf*)&As[r][c] = v;
    }
    { // stage B: coalesced bf16 copy, one column per thread
      const __bf16* bp = WT + (size_t)tid * HD + k0;
      #pragma unroll
      for (int j = 0; j < 8; ++j)
        *(v8bf*)&Bt[tid][j * 8] = *(const v8bf*)(bp + j * 8);
    }
    __syncthreads();
    #pragma unroll
    for (int kk = 0; kk < BK; kk += 32) {
      v16bf a0 = frag_a<LDT>(&As[0][kk],  lane);
      v16bf a1 = frag_a<LDT>(&As[16][kk], lane);
      v16bf b0 = frag_b<LDT>(&Bt[wave * 32][kk],      lane);
      v16bf b1 = frag_b<LDT>(&Bt[wave * 32 + 16][kk], lane);
      acc[0][0] = WMMA_BF16(a0, b0, acc[0][0]);
      acc[0][1] = WMMA_BF16(a0, b1, acc[0][1]);
      acc[1][0] = WMMA_BF16(a1, b0, acc[1][0]);
      acc[1][1] = WMMA_BF16(a1, b1, acc[1][1]);
    }
    __syncthreads();
  }

  #pragma unroll
  for (int rt = 0; rt < 2; ++rt)
    #pragma unroll
    for (int ct = 0; ct < 2; ++ct) {
      const int col = wave * 32 + ct * 16 + (lane & 15);
      const float bb = bias[col];
      #pragma unroll
      for (int i = 0; i < 8; ++i) {
        const int row = row0 + rt * 16 + (lane >> 4) * 8 + i;
        h2[(size_t)row * HD + col] = (__bf16)tanhf(acc[rt][ct][i] + bb);
      }
    }
}

// ---------------- Kernel 2: hwT = (h2 @ W_gcn)^T, bf16, stored [HD][NN] ----------------
__global__ __launch_bounds__(256)
void k_hw(const __bf16* __restrict__ h2, const __bf16* __restrict__ WT,
          __bf16* __restrict__ hwT) {
  __shared__ __bf16 As[BM][LDT];
  __shared__ __bf16 Bt[HD][LDT];
  const int tid = threadIdx.x, lane = tid & 31, wave = tid >> 5;
  const int row0 = blockIdx.x * BM;
  v8f acc[2][2] = {};

  for (int k0 = 0; k0 < HD; k0 += BK) {
    { // stage A: bf16 copy
      const int r = tid >> 3, c = (tid & 7) * 8;
      *(v8bf*)&As[r][c] = *(const v8bf*)(h2 + (size_t)(row0 + r) * HD + k0 + c);
    }
    { // stage B
      const __bf16* bp = WT + (size_t)tid * HD + k0;
      #pragma unroll
      for (int j = 0; j < 8; ++j)
        *(v8bf*)&Bt[tid][j * 8] = *(const v8bf*)(bp + j * 8);
    }
    __syncthreads();
    #pragma unroll
    for (int kk = 0; kk < BK; kk += 32) {
      v16bf a0 = frag_a<LDT>(&As[0][kk],  lane);
      v16bf a1 = frag_a<LDT>(&As[16][kk], lane);
      v16bf b0 = frag_b<LDT>(&Bt[wave * 32][kk],      lane);
      v16bf b1 = frag_b<LDT>(&Bt[wave * 32 + 16][kk], lane);
      acc[0][0] = WMMA_BF16(a0, b0, acc[0][0]);
      acc[0][1] = WMMA_BF16(a0, b1, acc[0][1]);
      acc[1][0] = WMMA_BF16(a1, b0, acc[1][0]);
      acc[1][1] = WMMA_BF16(a1, b1, acc[1][1]);
    }
    __syncthreads();
  }

  #pragma unroll
  for (int rt = 0; rt < 2; ++rt)
    #pragma unroll
    for (int ct = 0; ct < 2; ++ct) {
      const int col = wave * 32 + ct * 16 + (lane & 15);
      #pragma unroll
      for (int i = 0; i < 8; ++i) {
        const int row = row0 + rt * 16 + (lane >> 4) * 8 + i;
        hwT[(size_t)col * NN + row] = (__bf16)acc[rt][ct][i];  // transposed store
      }
    }
}

// ------- Kernel 3: fused filter + GEMM, double-buffered, async B staging -------
// out = relu(tanh(relu(adj*simlar)) @ hw + b_gcn)
__global__ __launch_bounds__(256)
void k_gcn(const float* __restrict__ adj, const float* __restrict__ simlar,
           const __bf16* __restrict__ hwT, const float* __restrict__ bias,
           float* __restrict__ out) {
  __shared__ __bf16 As[2][BM][LDT2];   //  6 KB
  __shared__ __bf16 Bt[2][HD][LDT2];   // 48 KB  (total 54 KB)
  const int tid = threadIdx.x, lane = tid & 31, wave = tid >> 5;
  const int row0 = blockIdx.x * BM;
  const int ar = tid >> 3;             // A-stage row   0..31
  const int ac = (tid & 7) * 4;        // A-stage col   0..28 (4 elems/thread)
  v8f acc[2][2] = {};

  // stage A: stream adj/simlar, filter elementwise, convert to bf16
  auto stageA = [&](int nb, int k0) {
    const size_t off = (size_t)(row0 + ar) * NN + k0 + ac;
    if (k0 + 2 * BK2 < NN) {           // hint the tile after next into cache
      __builtin_prefetch(adj + off + BK2, 0, 1);
      __builtin_prefetch(simlar + off + BK2, 0, 1);
    }
    const float4 a4 = *(const float4*)(adj + off);
    const float4 s4 = *(const float4*)(simlar + off);
    const float av[4] = {a4.x, a4.y, a4.z, a4.w};
    const float sv[4] = {s4.x, s4.y, s4.z, s4.w};
    v4bf v;
    #pragma unroll
    for (int j = 0; j < 4; ++j) {
      // top-k(T=700) with nnz<T keeps exactly the positive entries:
      // new_adj = tanh(max(adj*simlar, 0))   (adj >= 0)
      const float x = fmaxf(av[j] * sv[j], 0.0f);
      const float e = __expf(-2.0f * x);
      v[j] = (__bf16)((1.0f - e) / (1.0f + e));
    }
    *(v4bf*)&As[nb][ar][ac] = v;
  };

  // stage B: K-slab of hwT (L2-resident, 3 MB) straight into LDS
  auto stageB = [&](int nb, int k0) {
    const __bf16* bp = hwT + (size_t)tid * NN + k0;
    __bf16* dp = &Bt[nb][tid][0];
#if USE_ASYNC
    #pragma unroll
    for (int j = 0; j < 4; ++j)        // 16 B per lane per op, ASYNCcnt-tracked
      __builtin_amdgcn_global_load_async_to_lds_b128(
          GPTR(bp + j * 8), LPTR(dp + j * 8), 0, 0);
#else
    #pragma unroll
    for (int j = 0; j < 4; ++j)
      *(v8bf*)(dp + j * 8) = *(const v8bf*)(bp + j * 8);
#endif
  };

  stageA(0, 0);
  stageB(0, 0);
  int nb = 0;
  for (int it = 0; it < NN / BK2; ++it) {
#if USE_ASYNC
    __builtin_amdgcn_s_wait_asynccnt(0);   // __syncthreads fences DScnt only
#endif
    __syncthreads();                       // tile nb complete / prev reads done
    // issue fragment loads first so the WMMA dscnt-wait doesn't drain staging
    v16bf a0 = frag_a<LDT2>(&As[nb][0][0],  lane);
    v16bf a1 = frag_a<LDT2>(&As[nb][16][0], lane);
    v16bf b0 = frag_b<LDT2>(&Bt[nb][wave * 32][0],      lane);
    v16bf b1 = frag_b<LDT2>(&Bt[nb][wave * 32 + 16][0], lane);
    if (it + 1 < NN / BK2) {               // overlap next-tile staging with WMMA
      stageA(nb ^ 1, (it + 1) * BK2);
      stageB(nb ^ 1, (it + 1) * BK2);
    }
    acc[0][0] = WMMA_BF16(a0, b0, acc[0][0]);
    acc[0][1] = WMMA_BF16(a0, b1, acc[0][1]);
    acc[1][0] = WMMA_BF16(a1, b0, acc[1][0]);
    acc[1][1] = WMMA_BF16(a1, b1, acc[1][1]);
    nb ^= 1;
  }

  #pragma unroll
  for (int rt = 0; rt < 2; ++rt)
    #pragma unroll
    for (int ct = 0; ct < 2; ++ct) {
      const int col = wave * 32 + ct * 16 + (lane & 15);
      const float bb = bias[col];
      #pragma unroll
      for (int i = 0; i < 8; ++i) {
        const int row = row0 + rt * 16 + (lane >> 4) * 8 + i;
        out[(size_t)row * HD + col] = fmaxf(acc[rt][ct][i] + bb, 0.0f);
      }
    }
}

extern "C" void kernel_launch(void* const* d_in, const int* in_sizes, int n_in,
                              void* d_out, int out_size, void* d_ws, size_t ws_size,
                              hipStream_t stream) {
  const float* h      = (const float*)d_in[0];
  const float* adj    = (const float*)d_in[1];
  const float* simlar = (const float*)d_in[2];
  const float* Wp     = (const float*)d_in[3];
  const float* bp     = (const float*)d_in[4];
  const float* Wg     = (const float*)d_in[5];
  const float* bg     = (const float*)d_in[6];
  float* out = (float*)d_out;

  char* ws = (char*)d_ws;
  __bf16* WTp = (__bf16*)(ws);                                   // 128 KB
  __bf16* WTg = (__bf16*)(ws + (size_t)131072);                  // 128 KB
  __bf16* h2  = (__bf16*)(ws + (size_t)262144);                  // 3 MB
  __bf16* hwT = (__bf16*)(ws + (size_t)262144 + 3145728);        // 3 MB (total 6.25 MB)

  k_prep<<<dim3((HD * HD) / 256), dim3(256), 0, stream>>>(Wp, Wg, WTp, WTg);
  k_proj<<<dim3(NN / BM), dim3(256), 0, stream>>>(h, WTp, bp, h2);
  k_hw  <<<dim3(NN / BM), dim3(256), 0, stream>>>(h2, WTg, hwT);
  k_gcn <<<dim3(NN / BM), dim3(256), 0, stream>>>(adj, simlar, hwT, bg, out);
}
